// DecoderLayer_16922171146703
// MI455X (gfx1250) — matrix operand order repaired and checked by reading, not compiled, not run on hardware
//
#include <hip/hip_runtime.h>
#include <hip/hip_bf16.h>

// ---------------------------------------------------------------------------
// MI455X (gfx1250) decoder layer. Compute-bound (~322 GFLOP vs ~150MB HBM),
// so all matmuls go through V_WMMA_F32_16X16X32_BF16 (wave32), with LDS
// double-buffering fed by gfx1250 async-to-LDS loads + Tensor Data Mover.
// Block tile 128(M) x 64(N), K-step 32, 8 waves (4 over M x 2 over N),
// wave tile 32x32 = 4 WMMA accumulators (2 A frags x 2 B frags).
// Steady-state K-loop is peeled: unconditional stage/wait/compute, tail does
// the final wait-to-zero + compute (no per-iteration predicate logic).
// ---------------------------------------------------------------------------

typedef __attribute__((ext_vector_type(16))) __bf16       v16bf;
typedef __attribute__((ext_vector_type(8)))  float        v8f;
typedef __attribute__((ext_vector_type(8)))  unsigned int v8u;
typedef __attribute__((ext_vector_type(4)))  unsigned int u32x4;
typedef __attribute__((ext_vector_type(8)))  int          i32x8;
typedef __attribute__((ext_vector_type(4)))  int          i32x4;

union FragAB { v8u u; v16bf b; };

#define TM 128
#define TN 64
#define TK 32
#define LDS_PITCH (TK + 16)    // ushorts -> 96B row stride (16B-aligned vectors)

#if __has_builtin(__builtin_amdgcn_global_load_async_to_lds_b128)
#define HAVE_ASYNC_LDS 1
#else
#define HAVE_ASYNC_LDS 0
#endif

#if __has_builtin(__builtin_amdgcn_tensor_load_to_lds)
#define HAVE_TDM 1
#else
#define HAVE_TDM 0
#endif

// The async builtin takes typed v4i pointers in AS1 (global) / AS3 (LDS);
// C-style casts are the one form clang accepts for the addrspace conversion.
#define AS_G128(p) ((__attribute__((address_space(1))) i32x4*)(p))
#define AS_L128(p) ((__attribute__((address_space(3))) i32x4*)(p))

template<int N>
__device__ __forceinline__ void wait_async()
{
#if HAVE_ASYNC_LDS
#if __has_builtin(__builtin_amdgcn_s_wait_asynccnt)
    __builtin_amdgcn_s_wait_asynccnt(N);
#else
    if      (N == 0) asm volatile("s_wait_asynccnt 0x0" ::: "memory");
    else if (N == 2) asm volatile("s_wait_asynccnt 0x2" ::: "memory");
    else             asm volatile("s_wait_asynccnt 0x3" ::: "memory");
#endif
#endif
}

#if HAVE_TDM
// Issue one TDM 2-D tile load: tile_k contiguous elements per row (bf16),
// tile_n rows, global row stride = ld elements, into LDS with a 32B pad
// after every 64B row (matches LDS_PITCH). D# packing per ISA 8.3/8.4.
__device__ __forceinline__ void tdm_load_2d(unsigned lds_addr, const void* gaddr,
                                            int tile_k, int tile_n, long ld)
{
    unsigned long ga = (unsigned long)gaddr;
    u32x4 g0;
    g0[0] = 1u;                                               // count=1 (valid)
    g0[1] = lds_addr;                                         // lds_addr
    g0[2] = (unsigned)(ga & 0xffffffffu);                     // global_addr lo
    g0[3] = (unsigned)((ga >> 32) & 0x1ffffffu) | (2u << 30); // ga hi | type=2
    i32x8 g1;
    g1[0] = (1 << 16) |               // data_size = 2 bytes
            (1 << 20) |               // pad_enable
            (3 << 22) |               // pad_interval: every 16 dwords (64B row)
            (7 << 25);                // pad_amount: 8 dwords (32B)
    g1[1] = (tile_k & 0xffff) << 16;                          // tensor_dim0 lo
    g1[2] = ((unsigned)tile_k >> 16) | ((tile_n & 0xffff) << 16); // td0 hi|td1 lo
    g1[3] = ((unsigned)tile_n >> 16) | ((tile_k & 0xffff) << 16); // td1 hi|tile_dim0
    g1[4] = (tile_n & 0xffff);                                // tile_dim1; tile_dim2=0
    g1[5] = (int)(ld & 0xffffffffL);                          // dim0_stride lo
    g1[6] = (int)((ld >> 32) & 0xffffL);                      // dim0_stride hi
    g1[7] = 0;
    i32x4 z4 = {0, 0, 0, 0};
#if __has_include(<hip/amd_detail/amd_gfx1250_TDM.h>)
    i32x8 z8 = {0, 0, 0, 0, 0, 0, 0, 0};
    __builtin_amdgcn_tensor_load_to_lds(g0, g1, z4, z4, z8, 0);   // clang-23 form
#else
    __builtin_amdgcn_tensor_load_to_lds(g0, g1, z4, z4, 0);       // ROCm 7.2 form
#endif
}
#endif

// C[M,N] = act( scale * A[M,K] @ op(B) + bias ), batched over gridDim.z.
// BT=true : B stored [N,K] (K contiguous)  -> x @ W^T style (TDM-tiled B)
// BT=false: B stored [K,N] (N contiguous)  -> attn @ V style
// Per-operand batch offset: (z/nh)*s1 + (z%nh)*s2  (head slicing).
template<bool BT, bool RELU, typename OutT>
__global__ void __launch_bounds__(256)
gemm_wmma(const __bf16* __restrict__ A, const __bf16* __restrict__ Bm,
          const float* __restrict__ bias, OutT* __restrict__ C,
          int M, int N, int K,
          int nhA, long sA1, long sA2, int lda,
          int nhB, long sB1, long sB2, int ldb,
          int nhC, long sC1, long sC2, int ldc,
          float scale)
{
    (void)M; (void)N;
    const int z = blockIdx.z;
    const __bf16* Ab = A  + (long)(z / nhA) * sA1 + (long)(z % nhA) * sA2;
    const __bf16* Bb = Bm + (long)(z / nhB) * sB1 + (long)(z % nhB) * sB2;
    OutT*         Cb = C  + (long)(z / nhC) * sC1 + (long)(z % nhC) * sC2;

    const int m0   = blockIdx.y * TM;
    const int n0   = blockIdx.x * TN;
    const int tid  = threadIdx.x;
    const int wave = tid >> 5;
    const int lane = tid & 31;
    const int wm   = (wave & 3) * 32;    // 4 waves tile M=128
    const int wn   = (wave >> 2) * 32;   // 2 waves tile N=64

    __shared__ unsigned short As[2][TM][LDS_PITCH];   // double-buffered
    __shared__ unsigned short Bs[2][TN][LDS_PITCH];

    v8f acc[2][2];
#pragma unroll
    for (int i = 0; i < 2; ++i)
#pragma unroll
        for (int j = 0; j < 2; ++j)
#pragma unroll
            for (int r = 0; r < 8; ++r) acc[i][j][r] = 0.0f;

    // Stage tile (k0) into buffer p.
    auto stage = [&](int p, int k0) {
        // A tile (TM x TK): 512 x 16B, 2 per thread.
#pragma unroll
        for (int i = 0; i < 2; ++i) {
            int li  = tid + i * 256;
            int row = li >> 2;
            int cg  = (li & 3) << 3;
            const __bf16* src = Ab + (long)(m0 + row) * lda + k0 + cg;
#if HAVE_ASYNC_LDS
            __builtin_amdgcn_global_load_async_to_lds_b128(
                AS_G128(src), AS_L128(&As[p][row][cg]), 0, 0);
#else
            uint4 v = *(const uint4*)src;
            *(uint4*)&As[p][row][cg] = v;
#endif
        }
        if (BT) {
#if HAVE_TDM
            if (wave == 0)   // one TDM descriptor moves the whole B tile
                tdm_load_2d((unsigned)(size_t)
                                ((__attribute__((address_space(3))) unsigned short*)
                                     &Bs[p][0][0]),
                            Bb + (long)n0 * ldb + k0, TK, TN, (long)ldb);
#elif HAVE_ASYNC_LDS
            int row = tid >> 2, cg = (tid & 3) << 3;
            __builtin_amdgcn_global_load_async_to_lds_b128(
                AS_G128(Bb + (long)(n0 + row) * ldb + k0 + cg),
                AS_L128(&Bs[p][row][cg]), 0, 0);
#else
            int row = tid >> 2, cg = (tid & 3) << 3;
            uint4 v = *(const uint4*)(Bb + (long)(n0 + row) * ldb + k0 + cg);
            *(uint4*)&Bs[p][row][cg] = v;
#endif
        } else {
            // [K,N] -> [n][k] transpose through registers (~1.4% of total FLOPs).
            int kk = tid >> 3, ng = (tid & 7) << 3;
            uint4 v = *(const uint4*)(Bb + (long)(k0 + kk) * ldb + n0 + ng);
            const unsigned short* hs = (const unsigned short*)&v;
#pragma unroll
            for (int j = 0; j < 8; ++j) Bs[p][ng + j][kk] = hs[j];
        }
    };

    // Fragment loads (contiguous k = 16*(lane/16) + s, same for A and B) + 4 WMMA on buffer p.
    auto compute = [&](int p) {
        FragAB fa[2], fb[2];
        const int l15 = lane & 15;
        const int akb = (lane >> 4) << 4;   // A: lane hi half -> K base 16
        const int bkb = (lane >> 4) << 4;   // B: lane hi half -> K base 16
#pragma unroll
        for (int mt = 0; mt < 2; ++mt) {
            const int arow = wm + mt * 16 + l15;
#pragma unroll
            for (int v = 0; v < 8; ++v) {
                int k = akb + v * 2;
                fa[mt].u[v] = *(const unsigned int*)&As[p][arow][k];
            }
        }
#pragma unroll
        for (int nt = 0; nt < 2; ++nt) {
            const int bn = wn + nt * 16 + l15;
#pragma unroll
            for (int v = 0; v < 8; ++v)
                fb[nt].u[v] = *(const unsigned int*)&Bs[p][bn][bkb + v * 2];
        }
#pragma unroll
        for (int mt = 0; mt < 2; ++mt)
#pragma unroll
            for (int nt = 0; nt < 2; ++nt)
                acc[mt][nt] = __builtin_amdgcn_wmma_f32_16x16x32_bf16(
                    false, fa[mt].b, false, fb[nt].b, (short)0,
                    acc[mt][nt], false, false);
    };

    // Per-stage async issues by this thread: A(2) + B(1 if async B used).
    constexpr int PEND = (BT && !HAVE_TDM && HAVE_ASYNC_LDS) ? 3 : 2;

    stage(0, 0);
    int parity = 0;
    // ---- steady state: next tile always staged, unconditional waits ----
    for (int k0 = 0; k0 < K - TK; k0 += TK) {
        stage(parity ^ 1, k0 + TK);     // overlap next tile DMA with compute
        wait_async<PEND>();             // current tile's async loads done
        if (BT) {
#if HAVE_TDM
            if (wave == 0) __builtin_amdgcn_s_wait_tensorcnt(1);
#endif
        }
        __syncthreads();
        compute(parity);
        __syncthreads();                // all waves done before re-staging
        parity ^= 1;
    }
    // ---- tail: last tile, drain all DMA ----
    wait_async<0>();
    if (BT) {
#if HAVE_TDM
        if (wave == 0) __builtin_amdgcn_s_wait_tensorcnt(0);
#endif
    }
    __syncthreads();
    compute(parity);

    // ---- epilogue: C/D layout — lanes 0-15: M=r, lanes 16-31: M=r+8 ----
#pragma unroll
    for (int mt = 0; mt < 2; ++mt)
#pragma unroll
        for (int nt = 0; nt < 2; ++nt) {
            const int col = n0 + wn + nt * 16 + (lane & 15);
            const float bvv = bias ? bias[col] : 0.0f;
#pragma unroll
            for (int r = 0; r < 8; ++r) {
                const int row = m0 + wm + mt * 16 + ((lane < 16) ? r : r + 8);
                float val = acc[mt][nt][r] * scale + bvv;
                if (RELU) val = fmaxf(val, 0.0f);
                Cb[(long)row * ldc + col] = (OutT)val;
            }
        }
}

// f32 -> bf16 converter (grid-stride).
__global__ void cast_f32_bf16(const float* __restrict__ in,
                              __bf16* __restrict__ out, long n)
{
    long i = (long)blockIdx.x * blockDim.x + threadIdx.x;
    long stride = (long)gridDim.x * blockDim.x;
    for (; i < n; i += stride) out[i] = (__bf16)in[i];
}

// Row softmax over S=128 with int mask; emits bf16 probabilities.
__global__ void __launch_bounds__(128)
softmax128(const float* __restrict__ sc, const int* __restrict__ mask,
           __bf16* __restrict__ attn)
{
    const int S = 128;
    const long row = blockIdx.x;          // z*S + q
    const int  q   = (int)(row & (S - 1));
    const int  c   = threadIdx.x;

    float v = sc[row * S + c];
    if (mask[q * S + c] == 0) v = -1e9f;

    __shared__ float red[4];
    float m = v;
#pragma unroll
    for (int off = 16; off > 0; off >>= 1) m = fmaxf(m, __shfl_down(m, off, 32));
    if (!(threadIdx.x & 31)) red[threadIdx.x >> 5] = m;
    __syncthreads();
    m = fmaxf(fmaxf(red[0], red[1]), fmaxf(red[2], red[3]));

    float e = __expf(v - m);
    __syncthreads();
    float s = e;
#pragma unroll
    for (int off = 16; off > 0; off >>= 1) s += __shfl_down(s, off, 32);
    if (!(threadIdx.x & 31)) red[threadIdx.x >> 5] = s;
    __syncthreads();
    s = red[0] + red[1] + red[2] + red[3];

    attn[row * S + c] = (__bf16)(e / s);
}

// out = LN(x + y); optionally also emit bf16 copy for downstream GEMM A.
template<bool WB>
__global__ void __launch_bounds__(256)
add_layernorm(const float* __restrict__ x, const float* __restrict__ y,
              const float* __restrict__ w, const float* __restrict__ b,
              float* __restrict__ out, __bf16* __restrict__ outb, int D)
{
    const long row = blockIdx.x;
    const float* xr = x + row * (long)D;
    const float* yr = y + row * (long)D;

    float s = 0.0f, s2 = 0.0f;
    for (int c = threadIdx.x; c < D; c += 256) {
        float v = xr[c] + yr[c];
        s += v; s2 += v * v;
    }
    __shared__ float rs[8], rq[8];
#pragma unroll
    for (int off = 16; off > 0; off >>= 1) {
        s  += __shfl_down(s,  off, 32);
        s2 += __shfl_down(s2, off, 32);
    }
    const int lane = threadIdx.x & 31, wv = threadIdx.x >> 5;
    if (!lane) { rs[wv] = s; rq[wv] = s2; }
    __syncthreads();
    float S = 0.0f, S2 = 0.0f;
#pragma unroll
    for (int i = 0; i < 8; ++i) { S += rs[i]; S2 += rq[i]; }

    const float mean = S / (float)D;
    const float var  = S2 / (float)D - mean * mean;
    const float inv  = rsqrtf(var + 1e-12f);

    for (int c = threadIdx.x; c < D; c += 256) {
        float v = xr[c] + yr[c];
        float o = w[c] * (v - mean) * inv + b[c];
        out[row * (long)D + c] = o;
        if (WB) outb[row * (long)D + c] = (__bf16)o;
    }
}

// ---------------------------------------------------------------------------

extern "C" void kernel_launch(void* const* d_in, const int* in_sizes, int n_in,
                              void* d_out, int out_size, void* d_ws, size_t ws_size,
                              hipStream_t stream)
{
    (void)in_sizes; (void)n_in; (void)out_size; (void)ws_size;
    const int  Bn = 64, S = 128, D = 1024, H = 8, DK = 128, DFF = 2048, D4 = 4096;
    const long T  = (long)Bn * S;      // 8192 tokens
    const int  ZH = Bn * H;            // 512 (b,h) batches

    const float* x    = (const float*)d_in[0];
    const int*   mask = (const int*)  d_in[1];
    const float* Wq   = (const float*)d_in[2];  const float* bq = (const float*)d_in[3];
    const float* Wk   = (const float*)d_in[4];  const float* bk = (const float*)d_in[5];
    const float* Wv   = (const float*)d_in[6];  const float* bv = (const float*)d_in[7];
    const float* Wo   = (const float*)d_in[8];  const float* bo = (const float*)d_in[9];
    const float* ln1w = (const float*)d_in[10]; const float* ln1b = (const float*)d_in[11];
    const float* W1   = (const float*)d_in[12]; const float* b1 = (const float*)d_in[13];
    const float* W2   = (const float*)d_in[14]; const float* b2 = (const float*)d_in[15];
    const float* W3   = (const float*)d_in[16]; const float* b3 = (const float*)d_in[17];
    const float* ln2w = (const float*)d_in[18]; const float* ln2b = (const float*)d_in[19];

    char* p = (char*)d_ws;
    auto alloc = [&](long bytes) -> char* {
        char* r = p; p += (bytes + 255) & ~255L; return r;
    };
    __bf16* xb   = (__bf16*)alloc(T * D * 2);
    __bf16* wqb  = (__bf16*)alloc((long)D * D * 2);
    __bf16* wkb  = (__bf16*)alloc((long)D * D * 2);
    __bf16* wvb  = (__bf16*)alloc((long)D * D * 2);
    __bf16* wob  = (__bf16*)alloc((long)D * D * 2);
    __bf16* w1b  = (__bf16*)alloc((long)DFF * D * 2);
    __bf16* w2b  = (__bf16*)alloc((long)D4 * DFF * 2);
    __bf16* w3b  = (__bf16*)alloc((long)D * D4 * 2);
    __bf16* Qb   = (__bf16*)alloc(T * D * 2);
    __bf16* Kb   = (__bf16*)alloc(T * D * 2);
    __bf16* Vb   = (__bf16*)alloc(T * D * 2);
    float*  scF  = (float*) alloc((long)ZH * S * S * 4);
    __bf16* atB  = (__bf16*)alloc((long)ZH * S * S * 2);
    __bf16* avB  = (__bf16*)alloc((long)ZH * S * DK * 2);
    __bf16* ctxB = (__bf16*)alloc(T * D * 2);
    float*  aoF  = (float*) alloc(T * D * 4);
    float*  hF   = (float*) alloc(T * D * 4);
    __bf16* hB   = (__bf16*)alloc(T * D * 2);
    __bf16* f1B  = (__bf16*)alloc(T * DFF * 2);
    __bf16* f2B  = (__bf16*)alloc(T * D4 * 2);
    float*  f3F  = (float*) alloc(T * D * 4);

    auto castL = [&](const float* in, __bf16* out, long n) {
        long blocks = (n + 255) / 256; if (blocks > 2048) blocks = 2048;
        cast_f32_bf16<<<(int)blocks, 256, 0, stream>>>(in, out, n);
    };
    castL(x,  xb,  T * D);
    castL(Wq, wqb, (long)D * D);   castL(Wk, wkb, (long)D * D);
    castL(Wv, wvb, (long)D * D);   castL(Wo, wob, (long)D * D);
    castL(W1, w1b, (long)DFF * D);
    castL(W2, w2b, (long)D4 * DFF);
    castL(W3, w3b, (long)D * D4);

    const long SD = (long)S * D, SS = (long)S * S, SDK = (long)S * DK;
    const float invsq = 0.08838834764831845f;   // 1/sqrt(128)

    // QKV projections: [T,D] @ W^T + bias -> bf16
    gemm_wmma<true,false,__bf16><<<dim3(D/TN, (int)(T/TM), 1), 256, 0, stream>>>(
        xb, wqb, bq, Qb, (int)T, D, D, 1,0,0,D, 1,0,0,D, 1,0,0,D, 1.0f);
    gemm_wmma<true,false,__bf16><<<dim3(D/TN, (int)(T/TM), 1), 256, 0, stream>>>(
        xb, wkb, bk, Kb, (int)T, D, D, 1,0,0,D, 1,0,0,D, 1,0,0,D, 1.0f);
    gemm_wmma<true,false,__bf16><<<dim3(D/TN, (int)(T/TM), 1), 256, 0, stream>>>(
        xb, wvb, bv, Vb, (int)T, D, D, 1,0,0,D, 1,0,0,D, 1,0,0,D, 1.0f);

    // scores[z] = Q_h @ K_h^T / sqrt(DK)   (z = b*H + h, head slices via nh=H)
    gemm_wmma<true,false,float><<<dim3(S/TN, S/TM, ZH), 256, 0, stream>>>(
        Qb, Kb, nullptr, scF, S, S, DK,
        H, SD, DK, D,   H, SD, DK, D,   1, SS, 0, S,   invsq);

    softmax128<<<(int)((long)ZH * S), 128, 0, stream>>>(scF, mask, atB);

    // av[z] = attn[z] @ V_h      (B is [K,N] row-major within head slice)
    gemm_wmma<false,false,__bf16><<<dim3(DK/TN, S/TM, ZH), 256, 0, stream>>>(
        atB, Vb, nullptr, avB, S, DK, S,
        1, SS, 0, S,    H, SD, DK, D,   1, SDK, 0, DK,  1.0f);

    // ctx = av @ V (the DK==S quirk); output scattered straight to [T, D]
    gemm_wmma<false,false,__bf16><<<dim3(DK/TN, S/TM, ZH), 256, 0, stream>>>(
        avB, Vb, nullptr, ctxB, S, DK, S,
        1, SDK, 0, DK,  H, SD, DK, D,   H, SD, DK, D,   1.0f);

    // attn_out = ctx @ Wo^T + bo  -> f32
    gemm_wmma<true,false,float><<<dim3(D/TN, (int)(T/TM), 1), 256, 0, stream>>>(
        ctxB, wob, bo, aoF, (int)T, D, D, 1,0,0,D, 1,0,0,D, 1,0,0,D, 1.0f);

    // h = LN(x + attn_out)  (f32 + bf16 copy)
    add_layernorm<true><<<(int)T, 256, 0, stream>>>(x, aoF, ln1w, ln1b, hF, hB, D);

    // FFN
    gemm_wmma<true,true,__bf16><<<dim3(DFF/TN, (int)(T/TM), 1), 256, 0, stream>>>(
        hB, w1b, b1, f1B, (int)T, DFF, D, 1,0,0,D, 1,0,0,D, 1,0,0,DFF, 1.0f);
    gemm_wmma<true,true,__bf16><<<dim3(D4/TN, (int)(T/TM), 1), 256, 0, stream>>>(
        f1B, w2b, b2, f2B, (int)T, D4, DFF, 1,0,0,DFF, 1,0,0,DFF, 1,0,0,D4, 1.0f);
    gemm_wmma<true,false,float><<<dim3(D/TN, (int)(T/TM), 1), 256, 0, stream>>>(
        f2B, w3b, b3, f3F, (int)T, D, D4, 1,0,0,D4, 1,0,0,D4, 1,0,0,D, 1.0f);

    // out = LN(h + f)
    add_layernorm<false><<<(int)T, 256, 0, stream>>>(hF, f3F, ln2w, ln2b,
                                                     (float*)d_out, nullptr, D);
}
